// SiLUAttention_83915071030176
// MI455X (gfx1250) — matrix operand order repaired and checked by reading, not compiled, not run on hardware
//
#include <hip/hip_runtime.h>
#include <hip/hip_bf16.h>
#include <stdint.h>

// ---------------------------------------------------------------------------
// SiLU attention for MI455X (gfx1250): bf16 WMMA pipeline, TDM double-buffered.
//   x:[4,2048,1024] f32, w_qkuv:[1024,4096], w_out:[1024,1024], gamma/beta:[64]
// ---------------------------------------------------------------------------

typedef __bf16 bf16;
typedef __attribute__((ext_vector_type(16))) __bf16 v16bf;
typedef __attribute__((ext_vector_type(8)))  __bf16 v8bf;
typedef __attribute__((ext_vector_type(4)))  __bf16 v4bf;
typedef __attribute__((ext_vector_type(8)))  float  v8f;
typedef __attribute__((ext_vector_type(4)))  float  v4f;
typedef __attribute__((ext_vector_type(4)))  unsigned u32x4;
typedef __attribute__((ext_vector_type(8)))  int      i32x8;
typedef __attribute__((ext_vector_type(4)))  int      i32x4;

#define DIMD 1024
#define NH   16
#define HD   64
#define BB   4
#define LL   2048
#define M_TOT (BB*LL)            /* 8192 rows */
#define QSZ ((size_t)M_TOT*DIMD) /* elements per q/k/v/u plane */

// Fast SiLU: v_mul + v_exp_f32 + v_add + v_rcp_f32 + v_mul (no IEEE div expansion).
static __device__ __forceinline__ float silu_f(float x) {
  return x * __builtin_amdgcn_rcpf(1.0f + __expf(-x));
}

// ---- WMMA fragment helpers (CDNA5 16x16x32 bf16, per ISA 7.12.2 layouts) ---

static __device__ __forceinline__ v16bf cat16(v8bf lo, v8bf hi) {
  v16bf r;
#pragma unroll
  for (int i = 0; i < 8; ++i) { r[i] = lo[i]; r[i + 8] = hi[i]; }
  return r;
}

// A (16xK row-striped): lane<16 -> row=lane, K {0..7,16..23}; lane>=16 -> K {8..15,24..31}
static __device__ __forceinline__ v16bf load_a_frag(const bf16* base, int ldm, int lane) {
  const int row  = lane & 15;
  const int koff = (lane & 16) ? 8 : 0;
  const bf16* p = base + row * ldm + koff;
  return cat16(*(const v8bf*)p, *(const v8bf*)(p + 16));
}

// B (32x16): staged n-major in LDS: Bt[n][k]; lanes 0-15 hold K {0..7,16..23}, lanes 16-31 K {8..15,24..31}
static __device__ __forceinline__ v16bf load_b_frag(const bf16* base, int ldk, int lane) {
  const int col  = lane & 15;
  const int koff = (lane & 16) ? 8 : 0;
  const bf16* p = base + col * ldk + koff;
  return cat16(*(const v8bf*)p, *(const v8bf*)(p + 16));
}

static __device__ __forceinline__ v8f wmma_bf16f32(v16bf a, v16bf b, v8f c) {
  return __builtin_amdgcn_wmma_f32_16x16x32_bf16(false, a, false, b, (short)0, c,
                                                 false, false);
}

// ---- Tensor Data Mover: 2D tile (tile_d1 rows x tile_d0 elems) of bf16 -----

static __device__ __forceinline__ void wait_tensor0() {
#if __has_builtin(__builtin_amdgcn_s_wait_tensorcnt)
  __builtin_amdgcn_s_wait_tensorcnt(0);
#endif
}

#if __has_builtin(__builtin_amdgcn_tensor_load_to_lds)
#define HAVE_TDM 1
static __device__ __forceinline__ void tdm_load_tile_2d(
    unsigned lds_byte_off, const void* gptr,
    unsigned tensor_d0, unsigned tensor_d1, unsigned stride0,
    unsigned tile_d0, unsigned tile_d1) {
  uint64_t ga = (uint64_t)(uintptr_t)gptr;
  u32x4 g0;
  g0[0] = 1u;                                    // count=1, user descriptor
  g0[1] = lds_byte_off;                          // lds_addr [63:32]
  g0[2] = (unsigned)(ga & 0xFFFFFFFFu);          // global_addr lo
  g0[3] = (unsigned)((ga >> 32) & 0x01FFFFFFu) | (2u << 30); // addr hi | type=2
  i32x8 g1;
  g1[0] = (int)(1u << 16);                       // data_size=1 (2B), wg_mask=0
  g1[1] = (int)((tensor_d0 & 0xFFFFu) << 16);    // tensor_dim0[15:0] @ [63:48]
  g1[2] = (int)((tensor_d0 >> 16) & 0xFFFFu) | (int)((tensor_d1 & 0xFFFFu) << 16);
  g1[3] = (int)((tensor_d1 >> 16) & 0xFFFFu) | (int)((tile_d0 & 0xFFFFu) << 16);
  g1[4] = (int)(tile_d1 & 0xFFFFu);              // tile_dim1, tile_dim2=0
  g1[5] = (int)stride0;                          // tensor_dim0_stride lo
  g1[6] = 0;
  g1[7] = 0;
  i32x4 gz = {0, 0, 0, 0};
#if defined(__clang_major__) && (__clang_major__ >= 23)
  i32x8 gz8 = {0, 0, 0, 0, 0, 0, 0, 0};
  __builtin_amdgcn_tensor_load_to_lds(g0, g1, gz, gz, gz8, 0);
#else
  __builtin_amdgcn_tensor_load_to_lds(g0, g1, gz, gz, 0);
#endif
}
#else
#define HAVE_TDM 0
#endif

// ---------------------------------------------------------------------------
// fp32 -> bf16 convert pass (vectorized: float4 in, 4x bf16 out)
// ---------------------------------------------------------------------------
__global__ __launch_bounds__(256) void cvt_kernel(const float* __restrict__ in,
                                                  bf16* __restrict__ out, int n4) {
  int i = blockIdx.x * blockDim.x + threadIdx.x;
  int stride = gridDim.x * blockDim.x;
  for (; i < n4; i += stride) {
    v4f v = ((const v4f*)in)[i];
    v4bf o;
#pragma unroll
    for (int j = 0; j < 4; ++j) o[j] = (bf16)v[j];
    ((v4bf*)out)[i] = o;
  }
}

// ---------------------------------------------------------------------------
// Tiled bf16 GEMM: C[M=8192, NW] = A[8192,1024] x B[1024,NW]
//   MODE 0: silu epilogue, scatter to head-major q/k/v/u planes (bf16)
//   MODE 1: plain fp32 row-major output
// Block: 256 thr (8 waves), tile 128x128, K-step 32.
// Double-buffered LDS: A tile via TDM, B tile cooperatively staged n-major;
// tile i+1's DMA + global loads run under tile i's WMMAs.
// ---------------------------------------------------------------------------
template <int NW, int MODE>
__global__ __launch_bounds__(256) void gemm_kernel(const bf16* __restrict__ A,
                                                   const bf16* __restrict__ Bw,
                                                   void* __restrict__ outp) {
  __shared__ __align__(16) bf16 As[2][128 * 32];   // [m][k]
  __shared__ __align__(16) bf16 Bt[2][128 * 32];   // [n][k]
  const int t = threadIdx.x;
  const int lane = t & 31;
  const int wid = t >> 5;
  const int nb = blockIdx.x * 128;
  const int mb = blockIdx.y * 128;
  const int wm = wid >> 1;  // 0..3 -> 32-row strip
  const int wn = wid & 1;   // 0..1 -> 64-col strip
  const int K = 1024;

  v8f acc[2][4];
#pragma unroll
  for (int i = 0; i < 2; ++i)
#pragma unroll
    for (int j = 0; j < 4; ++j)
#pragma unroll
      for (int e = 0; e < 8; ++e) acc[i][j][e] = 0.0f;

  // Stage tile (A via TDM by wave 0, B cooperatively, transposed to n-major).
  auto stage = [&](int buf, int kt) {
#if HAVE_TDM
    if (wid == 0)
      tdm_load_tile_2d((unsigned)(uintptr_t)&As[buf][0], A + (size_t)mb * K + kt,
                       (unsigned)K, (unsigned)M_TOT, (unsigned)K, 32u, 128u);
#else
#pragma unroll
    for (int it = 0; it < 2; ++it) {
      int c = t + it * 256;            // 0..511 chunks of 8
      int row = c >> 2, kb0 = (c & 3) * 8;
      *(v8bf*)&As[buf][row * 32 + kb0] =
          *(const v8bf*)(A + (size_t)(mb + row) * K + kt + kb0);
    }
#endif
#pragma unroll
    for (int it = 0; it < 2; ++it) {
      int c = t + it * 256;            // 0..511
      int kk = c >> 4;                 // 0..31
      int n0 = (c & 15) * 8;
      v8bf v = *(const v8bf*)(Bw + (size_t)(kt + kk) * NW + nb + n0);
#pragma unroll
      for (int j = 0; j < 8; ++j) Bt[buf][(n0 + j) * 32 + kk] = v[j];
    }
  };

  stage(0, 0);
  int buf = 0;
  for (int kt = 0; kt < K; kt += 32, buf ^= 1) {
    if (wid == 0) wait_tensor0();   // TDM for `buf` complete (in-order TENSORcnt)
    __syncthreads();                // publish `buf` (barrier waits dscnt too)
    if (kt + 32 < K) stage(buf ^ 1, kt + 32);  // prefetch under the WMMAs

    v16bf af[2], bfr[4];
#pragma unroll
    for (int i = 0; i < 2; ++i)
      af[i] = load_a_frag(&As[buf][(wm * 32 + i * 16) * 32], 32, lane);
#pragma unroll
    for (int j = 0; j < 4; ++j)
      bfr[j] = load_b_frag(&Bt[buf][(wn * 64 + j * 16) * 32], 32, lane);
#pragma unroll
    for (int i = 0; i < 2; ++i)
#pragma unroll
      for (int j = 0; j < 4; ++j)
        acc[i][j] = wmma_bf16f32(af[i], bfr[j], acc[i][j]);
    __syncthreads();                // reads of `buf` done -> next stage may overwrite
  }

  // Epilogue. C layout: elem(r,lane): row = r + 8*(lane>=16), col = lane%16.
  const int rhalf = (lane & 16) ? 8 : 0;
  const int cl = lane & 15;
#pragma unroll
  for (int i = 0; i < 2; ++i)
#pragma unroll
    for (int j = 0; j < 4; ++j)
#pragma unroll
      for (int r = 0; r < 8; ++r) {
        int m = mb + wm * 32 + i * 16 + r + rhalf;
        int n = nb + wn * 64 + j * 16 + cl;
        float v = acc[i][j][r];
        if (MODE == 0) {
          bf16* qkuv = (bf16*)outp;
          v = silu_f(v);
          int which = n >> 10;               // 0:q 1:k 2:v 3:u
          int c1 = n & 1023;
          int h = c1 >> 6, dc = c1 & 63;
          int b = m >> 11, l = m & 2047;
          size_t idx = (size_t)which * QSZ +
                       (((size_t)(b * NH + h) * LL + l) * HD + dc);
          qkuv[idx] = (bf16)v;
        } else {
          float* O = (float*)outp;
          O[(size_t)m * NW + n] = v;
        }
      }
}

// ---------------------------------------------------------------------------
// Causal SiLU attention + LayerNorm*u, per (b,h), 128 q-rows per block.
// No softmax => plain streaming accumulation over causal key tiles.
// Q/K tiles are contiguous 16KB blocks -> moved by TDM; V is staged transposed
// by the waves (needs element shuffling), overlapping the K-tile DMA.
// ---------------------------------------------------------------------------
__global__ __launch_bounds__(256) void attn_kernel(const bf16* __restrict__ qkuv,
                                                   const float* __restrict__ gamma,
                                                   const float* __restrict__ beta,
                                                   bf16* __restrict__ attn_out) {
  extern __shared__ __align__(16) unsigned char smem[];
  bf16* Qs = (bf16*)smem;                  // [128][64]  16 KB
  bf16* Ks = (bf16*)(smem + 16384);        // [128][64]  16 KB (== Bt for QK^T)
  bf16* Vt = (bf16*)(smem + 32768);        // [64][128]  16 KB (n-major V)
  bf16* Ss = (bf16*)(smem + 49152);        // 8 x [16][128] bf16, 32 KB
  float* Os = (float*)(smem + 16384);      // LN spill, aliases Ks+Vt (32 KB)

  const int t = threadIdx.x, lane = t & 31, wid = t >> 5;
  const int qb = blockIdx.x * 128;
  const int h = blockIdx.y, b = blockIdx.z;
  const size_t head_off = ((size_t)(b * NH + h)) * LL * HD;
  const bf16* Qg = qkuv + head_off;
  const bf16* Kg = qkuv + QSZ + head_off;
  const bf16* Vg = qkuv + 2 * QSZ + head_off;
  const bf16* Ug = qkuv + 3 * QSZ + head_off;

  // Q tile: one TDM (completion folded into the first in-loop tensorcnt wait)
#if HAVE_TDM
  if (wid == 0)
    tdm_load_tile_2d((unsigned)(uintptr_t)Qs, Qg + (size_t)qb * HD,
                     (unsigned)HD, (unsigned)LL, (unsigned)HD, 64u, 128u);
#else
#pragma unroll
  for (int it = 0; it < 4; ++it) {
    int c = t + it * 256;
    int row = c >> 3, d0 = (c & 7) * 8;
    *(v8bf*)&Qs[row * 64 + d0] = *(const v8bf*)(Qg + (size_t)(qb + row) * HD + d0);
  }
#endif

  v8f oacc[4];
#pragma unroll
  for (int j = 0; j < 4; ++j)
#pragma unroll
    for (int e = 0; e < 8; ++e) oacc[j][e] = 0.0f;

  const float scale = 0.125f;  // 1/sqrt(64)
  bf16* Sp = Ss + wid * 16 * 128;
  const int rhalf = (lane & 16) ? 8 : 0;
  const int cl = lane & 15;

  for (int kb = 0; kb <= qb; kb += 128) {
    __syncthreads();  // previous iteration's reads of Ks/Vt complete
#if HAVE_TDM
    if (wid == 0)
      tdm_load_tile_2d((unsigned)(uintptr_t)Ks, Kg + (size_t)kb * HD,
                       (unsigned)HD, (unsigned)LL, (unsigned)HD, 64u, 128u);
#endif
    // V transposed to Vt[d][key] (and K manually if no TDM), under the DMA
#pragma unroll
    for (int it = 0; it < 4; ++it) {
      int c = t + it * 256;
      int row = c >> 3, d0 = (c & 7) * 8;
#if !HAVE_TDM
      *(v8bf*)&Ks[row * 64 + d0] =
          *(const v8bf*)(Kg + (size_t)(kb + row) * HD + d0);
#endif
      v8bf vv = *(const v8bf*)(Vg + (size_t)(kb + row) * HD + d0);
#pragma unroll
      for (int j = 0; j < 8; ++j) Vt[(d0 + j) * 128 + row] = vv[j];
    }
#if HAVE_TDM
    if (wid == 0) wait_tensor0();
#endif
    __syncthreads();

    // S = mask * silu(scale * Q K^T): wave computes 16 q-rows x 128 keys
    v16bf aq0 = load_a_frag(&Qs[(wid * 16) * 64], 64, lane);
    v16bf aq1 = load_a_frag(&Qs[(wid * 16) * 64 + 32], 64, lane);
#pragma unroll
    for (int nt = 0; nt < 8; ++nt) {
      v8f s;
#pragma unroll
      for (int e = 0; e < 8; ++e) s[e] = 0.0f;
      v16bf bk0 = load_b_frag(&Ks[(nt * 16) * 64], 64, lane);
      v16bf bk1 = load_b_frag(&Ks[(nt * 16) * 64 + 32], 64, lane);
      s = wmma_bf16f32(aq0, bk0, s);
      s = wmma_bf16f32(aq1, bk1, s);
      int colg = kb + nt * 16 + cl;
#pragma unroll
      for (int r = 0; r < 8; ++r) {
        int rowg = qb + wid * 16 + r + rhalf;
        float v = silu_f(s[r] * scale);
        v = (colg <= rowg) ? v : 0.0f;  // multiplicative causal mask
        Sp[(r + rhalf) * 128 + nt * 16 + cl] = (bf16)v;
      }
    }
    // O += S @ V  (A = Sp 16x128, B = Vt n-major)
#pragma unroll
    for (int dt = 0; dt < 4; ++dt)
#pragma unroll
      for (int kc = 0; kc < 4; ++kc) {
        v16bf a = load_a_frag(Sp + kc * 32, 128, lane);
        v16bf bb = load_b_frag(&Vt[(dt * 16) * 128 + kc * 32], 128, lane);
        oacc[dt] = wmma_bf16f32(a, bb, oacc[dt]);
      }
  }

  __syncthreads();  // done with Ks/Vt -> reuse as LN spill
  float* Op = Os + wid * 16 * 64;
#pragma unroll
  for (int dt = 0; dt < 4; ++dt)
#pragma unroll
    for (int r = 0; r < 8; ++r)
      Op[(r + rhalf) * 64 + dt * 16 + cl] = oacc[dt][r];
  __syncthreads();

  if (lane < 16) {  // one lane per q-row: LN over d=64, * u, store bf16
    int row = lane;
    int qrow = qb + wid * 16 + row;
    const float* pr = Op + row * 64;
    float mu = 0.0f;
#pragma unroll
    for (int c = 0; c < 64; ++c) mu += pr[c];
    mu *= (1.0f / 64.0f);
    float var = 0.0f;
#pragma unroll
    for (int c = 0; c < 64; ++c) { float d = pr[c] - mu; var += d * d; }
    var *= (1.0f / 64.0f);
    float rs = __builtin_amdgcn_rsqf(var + 1e-5f);  // single v_rsq_f32
    const bf16* up = Ug + (size_t)qrow * HD;
    bf16* op = attn_out + ((size_t)(b * LL + qrow)) * DIMD + h * HD;
#pragma unroll
    for (int c = 0; c < 64; ++c) {
      float y = (pr[c] - mu) * rs * gamma[c] + beta[c];
      y *= (float)up[c];
      op[c] = (bf16)y;
    }
  }
}

// ---------------------------------------------------------------------------
extern "C" void kernel_launch(void* const* d_in, const int* in_sizes, int n_in,
                              void* d_out, int out_size, void* d_ws, size_t ws_size,
                              hipStream_t stream) {
  const float* x      = (const float*)d_in[0];
  const float* w_qkuv = (const float*)d_in[1];
  const float* w_out  = (const float*)d_in[2];
  const float* gamma  = (const float*)d_in[3];
  const float* beta   = (const float*)d_in[4];
  // d_in[5] (attn_mask) unused: causal tril applied analytically.

  unsigned char* ws = (unsigned char*)d_ws;
  bf16* xb   = (bf16*)(ws);                              // 16 MB
  bf16* wqb  = (bf16*)(ws + (size_t)16 * 1024 * 1024);   //  8 MB
  bf16* wob  = (bf16*)(ws + (size_t)24 * 1024 * 1024);   //  2 MB
  bf16* qkuv = (bf16*)(ws + (size_t)26 * 1024 * 1024);   // 64 MB (q,k,v,u planes)
  bf16* attn = (bf16*)(ws + (size_t)90 * 1024 * 1024);   // 16 MB

  cvt_kernel<<<1024, 256, 0, stream>>>(x, xb, (M_TOT * DIMD) / 4);
  cvt_kernel<<<1024, 256, 0, stream>>>(w_qkuv, wqb, (DIMD * 4 * DIMD) / 4);
  cvt_kernel<<<512, 256, 0, stream>>>(w_out, wob, (DIMD * DIMD) / 4);

  gemm_kernel<4096, 0><<<dim3(32, 64), 256, 0, stream>>>(xb, wqb, (void*)qkuv);
  attn_kernel<<<dim3(LL / 128, NH, BB), 256, 80 * 1024, stream>>>(qkuv, gamma, beta,
                                                                  attn);
  gemm_kernel<1024, 1><<<dim3(8, 64), 256, 0, stream>>>(attn, wob, d_out);
}
